// GNN_3_7275674599611
// MI455X (gfx1250) — matrix operand order repaired and checked by reading, not compiled, not run on hardware
//
#include <hip/hip_runtime.h>
#include <hip/hip_bf16.h>
#include <stdint.h>

typedef __attribute__((ext_vector_type(16))) __bf16 v16bf;
typedef __attribute__((ext_vector_type(8)))  float  v8f;
typedef __attribute__((ext_vector_type(4)))  float  v4f;

// round-to-nearest bf16 bits of a float
__device__ __forceinline__ unsigned short bfr(float x) {
    unsigned u = __float_as_uint(x);
    return (unsigned short)((u + 0x7FFFu + ((u >> 16) & 1u)) >> 16);
}
__device__ __forceinline__ __bf16 bits2bf(unsigned short h) {
    return __builtin_bit_cast(__bf16, h);
}
__device__ __forceinline__ float bf2f(unsigned short h) {
    return __uint_as_float((unsigned)h << 16);
}

// ---------------- degree / normalization ----------------
__global__ void deg_init(float* deg, int n) {
    int i = blockIdx.x * blockDim.x + threadIdx.x;
    if (i < n) deg[i] = 1.0f;              // self-loop contributes 1
}
__global__ void deg_count(const int* __restrict__ dst, float* deg, int e) {
    int i = blockIdx.x * blockDim.x + threadIdx.x;
    if (i < e) atomicAdd(&deg[dst[i]], 1.0f);
}
__global__ void deg_rsqrt(float* deg, int n) {
    int i = blockIdx.x * blockDim.x + threadIdx.x;
    if (i < n) deg[i] = rsqrtf(deg[i]);    // deg >= 1 always (self-loops)
}

// ---------------- weight split: W[K][C] fp32 -> Whi/Wlo [C][K] bf16 ----------------
__global__ void w_split_t(const float* __restrict__ W, __bf16* __restrict__ hi,
                          __bf16* __restrict__ lo, int K, int C) {
    int idx = blockIdx.x * blockDim.x + threadIdx.x;
    if (idx >= K * C) return;
    int k = idx / C, c = idx - k * C;
    float w = W[idx];
    unsigned short h = bfr(w);
    hi[(size_t)c * K + k] = bits2bf(h);
    lo[(size_t)c * K + k] = bits2bf(bfr(w - bf2f(h)));
}

// ---------------- dense transform H = (relu?)X @ W via WMMA ----------------
// One wave = one 16x16 tile of H; one block = one 16-row stripe across all C.
// fp32 accuracy via 3x bf16 split-GEMM (hi*hi + hi*lo + lo*hi, f32 accum).
// Requires N % 16 == 0 (holds: N = 100000).
template <int K, int C, int RELU>
__global__ void gcn_gemm_wmma(const float* __restrict__ X,
                              const __bf16* __restrict__ Whi,
                              const __bf16* __restrict__ Wlo,
                              float* __restrict__ H) {
    __shared__ float stripe[16 * C];
    const int wave = threadIdx.x >> 5;
    const int lane = threadIdx.x & 31;
    const int half = lane >> 4;     // 0: lanes 0-15, 1: lanes 16-31
    const int l16  = lane & 15;
    const int r0   = blockIdx.x << 4;
    const int bcol = (wave << 4) + l16;

    const float*  __restrict__ xrow = X + (size_t)(r0 + l16) * K;
    const __bf16* __restrict__ whc  = Whi + (size_t)bcol * K;
    const __bf16* __restrict__ wlc  = Wlo + (size_t)bcol * K;

    v8f acc = {};
#pragma unroll
    for (int kt = 0; kt < K; kt += 32) {
        // A fragment: 16 contiguous floats along K per lane (k = 16*half + s, matches B)
        v4f a0 = *(const v4f*)(xrow + kt + half * 16);
        v4f a1 = *(const v4f*)(xrow + kt + half * 16 + 4);
        v4f a2 = *(const v4f*)(xrow + kt + half * 16 + 8);
        v4f a3 = *(const v4f*)(xrow + kt + half * 16 + 12);
        float av[16];
#pragma unroll
        for (int j = 0; j < 4; ++j) {
            av[j] = a0[j]; av[4 + j] = a1[j]; av[8 + j] = a2[j]; av[12 + j] = a3[j];
        }
        v16bf ahi, alo;
#pragma unroll
        for (int j = 0; j < 16; ++j) {
            float a = av[j];
            if (RELU) a = fmaxf(a, 0.0f);
            unsigned short h = bfr(a);
            ahi[j] = bits2bf(h);
            alo[j] = bits2bf(bfr(a - bf2f(h)));
        }
        // B fragment: 16 contiguous bf16 along K (pre-split, transposed)
        v16bf bhi = *(const v16bf*)(whc + kt + half * 16);
        v16bf blo = *(const v16bf*)(wlc + kt + half * 16);

        acc = __builtin_amdgcn_wmma_f32_16x16x32_bf16(false, ahi, false, bhi,
                                                      (short)0, acc, false, false);
        acc = __builtin_amdgcn_wmma_f32_16x16x32_bf16(false, ahi, false, blo,
                                                      (short)0, acc, false, false);
        acc = __builtin_amdgcn_wmma_f32_16x16x32_bf16(false, alo, false, bhi,
                                                      (short)0, acc, false, false);
    }
    // Stage tile in LDS, then block-coalesced b128 stores (stripe is contiguous in H)
#pragma unroll
    for (int r = 0; r < 8; ++r)
        stripe[(half * 8 + r) * C + bcol] = acc[r];
    __syncthreads();
    v4f* __restrict__ g = (v4f*)(H + (size_t)r0 * C);
    const v4f* __restrict__ s4 = (const v4f*)stripe;
    constexpr int TOTAL4 = 16 * C / 4;
    for (int i = threadIdx.x; i < TOTAL4; i += blockDim.x)
        g[i] = s4[i];
}

// ---------------- aggregation ----------------
// Out[n][c] = dinv[n]^2 * H[n][c] + bias[c]   (self-loop + bias folded in)
__global__ void agg_init4(const v4f* __restrict__ H, const float* __restrict__ dinv,
                          const v4f* __restrict__ bias, v4f* __restrict__ Out,
                          int total4, int C4) {
    int i = blockIdx.x * blockDim.x + threadIdx.x;
    if (i >= total4) return;
    int n = i / C4;
    float dd = dinv[n]; dd *= dd;
    v4f h = H[i], b = bias[i - n * C4];
    v4f o; o[0] = dd * h[0] + b[0]; o[1] = dd * h[1] + b[1];
    o[2] = dd * h[2] + b[2]; o[3] = dd * h[3] + b[3];
    Out[i] = o;
}

// One wave per edge: Out[dst] += dinv[src]*dinv[dst] * H[src]
template <int C>
__global__ void agg_edges(const float* __restrict__ H,
                          const int* __restrict__ src, const int* __restrict__ dst,
                          const float* __restrict__ dinv, float* __restrict__ Out,
                          int E) {
    int gw = (blockIdx.x * blockDim.x + threadIdx.x) >> 5;
    int lane = threadIdx.x & 31;
    if (gw >= E) return;
    int s = src[gw], d = dst[gw];
    float w = dinv[s] * dinv[d];
    constexpr int V = C / 32;      // floats per lane (4 or 2)
    const float* __restrict__ hs = H + (size_t)s * C + lane * V;
    float* od = Out + (size_t)d * C + lane * V;
    if constexpr (V == 4) {
        v4f h4 = *(const v4f*)hs;
        atomicAdd(od + 0, w * h4[0]); atomicAdd(od + 1, w * h4[1]);
        atomicAdd(od + 2, w * h4[2]); atomicAdd(od + 3, w * h4[3]);
    } else {
        float2 h2 = *(const float2*)hs;
        atomicAdd(od + 0, w * h2.x); atomicAdd(od + 1, w * h2.y);
    }
}

extern "C" void kernel_launch(void* const* d_in, const int* in_sizes, int n_in,
                              void* d_out, int out_size, void* d_ws, size_t ws_size,
                              hipStream_t stream) {
    (void)n_in; (void)out_size; (void)ws_size;
    const float* x  = (const float*)d_in[0];
    const int*   ei = (const int*)d_in[1];
    const float* W1 = (const float*)d_in[2];
    const float* b1 = (const float*)d_in[3];
    const float* W2 = (const float*)d_in[4];
    const float* b2 = (const float*)d_in[5];
    const float* W3 = (const float*)d_in[6];
    const float* b3 = (const float*)d_in[7];

    const int IN_C = 128, HID = 128, OUT = 64;
    const int N = in_sizes[0] / IN_C;
    const int E = in_sizes[1] / 2;
    const int* srcp = ei;          // edge_index[0]
    const int* dstp = ei + E;      // edge_index[1]

    float*  dinv = (float*)d_ws;                        // N
    float*  buf0 = dinv + N;                            // N*HID (gemm out)
    float*  buf1 = buf0 + (size_t)N * HID;              // N*HID (agg out)
    __bf16* whi  = (__bf16*)(buf1 + (size_t)N * HID);   // 128*128 bf16
    __bf16* wlo  = whi + 128 * 128;                     // 128*128 bf16
    float*  outp = (float*)d_out;                       // N*OUT

    const int T = 256;
    // normalization: dinv = rsqrt(1 + in-degree)
    deg_init <<<(N + T - 1) / T, T, 0, stream>>>(dinv, N);
    deg_count<<<(E + T - 1) / T, T, 0, stream>>>(dstp, dinv, E);
    deg_rsqrt<<<(N + T - 1) / T, T, 0, stream>>>(dinv, N);

    const int rowTiles = N / 16;                        // N % 16 == 0
    const int ewBlocks = (E + 7) / 8;                   // 1 wave / edge

    // Layer 1: H1 = X@W1 ; A1 = agg(H1) + b1   (relu folded into next gemm)
    w_split_t<<<(128 * 128 + T - 1) / T, T, 0, stream>>>(W1, whi, wlo, 128, 128);
    gcn_gemm_wmma<128, 128, 0><<<rowTiles, 256, 0, stream>>>(x, whi, wlo, buf0);
    agg_init4<<<(N * HID / 4 + T - 1) / T, T, 0, stream>>>(
        (const v4f*)buf0, dinv, (const v4f*)b1, (v4f*)buf1, N * HID / 4, HID / 4);
    agg_edges<128><<<ewBlocks, T, 0, stream>>>(buf0, srcp, dstp, dinv, buf1, E);

    // Layer 2: H2 = relu(A1)@W2 ; A2 = agg(H2) + b2
    w_split_t<<<(128 * 128 + T - 1) / T, T, 0, stream>>>(W2, whi, wlo, 128, 128);
    gcn_gemm_wmma<128, 128, 1><<<rowTiles, 256, 0, stream>>>(buf1, whi, wlo, buf0);
    agg_init4<<<(N * HID / 4 + T - 1) / T, T, 0, stream>>>(
        (const v4f*)buf0, dinv, (const v4f*)b2, (v4f*)buf1, N * HID / 4, HID / 4);
    agg_edges<128><<<ewBlocks, T, 0, stream>>>(buf0, srcp, dstp, dinv, buf1, E);

    // Layer 3: H3 = relu(A2)@W3 ; out = agg(H3) + b3
    w_split_t<<<(128 * 64 + T - 1) / T, T, 0, stream>>>(W3, whi, wlo, 128, 64);
    gcn_gemm_wmma<128, 64, 1><<<rowTiles, 128, 0, stream>>>(buf1, whi, wlo, buf0);
    agg_init4<<<(N * OUT / 4 + T - 1) / T, T, 0, stream>>>(
        (const v4f*)buf0, dinv, (const v4f*)b3, (v4f*)outp, N * OUT / 4, OUT / 4);
    agg_edges<64><<<ewBlocks, T, 0, stream>>>(buf0, srcp, dstp, dinv, outp, E);
}
